// LSTM_71382356459716
// MI455X (gfx1250) — matrix operand order repaired and checked by reading, not compiled, not run on hardware
//
#include <hip/hip_runtime.h>

#define T_STEPS 28
#define INPUT_N 28
#define HID 64
#define OUT_N 10
#define ROWS_PER_WAVE 16
#define WAVES_PER_BLOCK 8
#define ROWS_PER_BLOCK (ROWS_PER_WAVE * WAVES_PER_BLOCK)

// LDS layout (bytes)
#define SW_IH_OFF   0                    // 256 rows * 40 bf16 (K padded to 32, stride 80B)
#define SW_HH_OFF   20480                // 256 rows * 72 bf16 (K=64, stride 144B)
#define SBIAS_OFF   57344                // 256 floats
#define SH_OFF      58368                // 8 waves * 16 rows * 72 bf16 (stride 144B)
#define SMEM_BYTES  76800

typedef __attribute__((ext_vector_type(16))) __bf16 v16bf;
typedef __attribute__((ext_vector_type(8)))  float  v8f;

union BF16Frag {
    __bf16 b[16];
    uint4  q[2];
    v16bf  v;
};

// Native V_TANH_F32 on gfx1250 (TRANS pipe, co-executes with the XDL/WMMA pipe).
#if __has_builtin(__builtin_amdgcn_tanhf)
__device__ __forceinline__ float tanhx(float x) { return __builtin_amdgcn_tanhf(x); }
#else
__device__ __forceinline__ float tanhx(float x) {
    return 1.0f - 2.0f * __builtin_amdgcn_rcpf(__expf(2.0f * x) + 1.0f);
}
#endif
__device__ __forceinline__ float sigm(float x) { return 0.5f * tanhx(0.5f * x) + 0.5f; }

__global__ __launch_bounds__(256) void lstm_fused_wmma(
    const float* __restrict__ x,      // [B, 28, 28]
    const float* __restrict__ W_ih,   // [256, 28]
    const float* __restrict__ W_hh,   // [256, 64]
    const float* __restrict__ b_ih,   // [256]
    const float* __restrict__ b_hh,   // [256]
    const float* __restrict__ W_fc,   // [10, 64]
    const float* __restrict__ b_fc,   // [10]
    float* __restrict__ out)          // [B, 10]
{
    extern __shared__ char smem[];
    __bf16* sWih  = (__bf16*)(smem + SW_IH_OFF);
    __bf16* sWhh  = (__bf16*)(smem + SW_HH_OFF);
    float*  sBias = (float*)(smem + SBIAS_OFF);
    __bf16* sH    = (__bf16*)(smem + SH_OFF);

    const int tid  = threadIdx.x;
    const int wave = tid >> 5;
    const int lane = tid & 31;
    const int lh   = lane >> 4;   // lane half (0/1) -> K sub-block select
    const int mrow = lane & 15;   // M row (A frag) / N column (B,C,D frags)

    // ---- stage weights to LDS as bf16 (hardware cvt), pre-sum bias ----
    {
        const int r = tid;  // 256 threads == 4*HID rows
        const float* srci = W_ih + r * INPUT_N;
        __bf16* dsti = sWih + r * 40;
        #pragma unroll
        for (int k = 0; k < INPUT_N; ++k) dsti[k] = (__bf16)srci[k];
        #pragma unroll
        for (int k = INPUT_N; k < 32; ++k) dsti[k] = (__bf16)0.0f;  // pad K 28->32

        const float* srch = W_hh + r * HID;
        __bf16* dsth = sWhh + r * 72;
        #pragma unroll
        for (int k = 0; k < HID; ++k) dsth[k] = (__bf16)srch[k];

        sBias[r] = b_ih[r] + b_hh[r];
    }
    __syncthreads();

    // per-wave private h tile in LDS (16 rows x stride 72 bf16); h0 = 0
    __bf16* sHw = sH + wave * (16 * 72);
    for (int i = lane; i < 16 * 72; i += 32) sHw[i] = (__bf16)0.0f;

    // hoist the per-lane bias for all 16 gate tiles out of the t-loop
    float bv[16];
    #pragma unroll
    for (int g = 0; g < 4; ++g)
        #pragma unroll
        for (int j = 0; j < 4; ++j)
            bv[g * 4 + j] = sBias[g * 64 + j * 16 + mrow];

    const int b0 = blockIdx.x * ROWS_PER_BLOCK + wave * ROWS_PER_WAVE;
    const float* xrow = x + (size_t)(b0 + mrow) * (T_STEPS * INPUT_N);

    v8f cc[4];
    #pragma unroll
    for (int j = 0; j < 4; ++j) cc[j] = (v8f){0.f,0.f,0.f,0.f,0.f,0.f,0.f,0.f};

    for (int t = 0; t < T_STEPS; ++t) {
        // ---- build x A-fragment (16x32 bf16, K 28..31 zero) straight from HBM ----
        BF16Frag ax;
        {
            const float* xp = xrow + t * INPUT_N;
            const int k1 = lh * 8;
            float4 a0 = *(const float4*)(xp + k1);
            float4 a1 = *(const float4*)(xp + k1 + 4);
            float4 c0v, c1v;
            if (lh == 0) {
                c0v = *(const float4*)(xp + 16);
                c1v = *(const float4*)(xp + 20);
            } else {
                c0v = *(const float4*)(xp + 24);
                c1v.x = 0.f; c1v.y = 0.f; c1v.z = 0.f; c1v.w = 0.f;  // K = 28..31
            }
            ax.b[0]  = (__bf16)a0.x;  ax.b[1]  = (__bf16)a0.y;
            ax.b[2]  = (__bf16)a0.z;  ax.b[3]  = (__bf16)a0.w;
            ax.b[4]  = (__bf16)a1.x;  ax.b[5]  = (__bf16)a1.y;
            ax.b[6]  = (__bf16)a1.z;  ax.b[7]  = (__bf16)a1.w;
            ax.b[8]  = (__bf16)c0v.x; ax.b[9]  = (__bf16)c0v.y;
            ax.b[10] = (__bf16)c0v.z; ax.b[11] = (__bf16)c0v.w;
            ax.b[12] = (__bf16)c1v.x; ax.b[13] = (__bf16)c1v.y;
            ax.b[14] = (__bf16)c1v.z; ax.b[15] = (__bf16)c1v.w;
            if (t + 1 < T_STEPS) __builtin_prefetch(xp + INPUT_N, 0, 0);
        }

        // ---- h A-fragments for K=0..31 and K=32..63 from per-wave LDS tile ----
        BF16Frag ah0, ah1;
        {
            const char* hr = (const char*)(sHw + mrow * 72);
            ah0.q[0] = *(const uint4*)(hr + lh * 16);
            ah0.q[1] = *(const uint4*)(hr + lh * 16 + 32);
            ah1.q[0] = *(const uint4*)(hr + 64 + lh * 16);
            ah1.q[1] = *(const uint4*)(hr + 64 + lh * 16 + 32);
        }

        #pragma unroll
        for (int j = 0; j < 4; ++j) {
            v8f ga[4];
            #pragma unroll
            for (int g = 0; g < 4; ++g) {
                const int n = g * 64 + j * 16 + mrow;   // gate column this lane owns

                // issue all six ds_load_b128 for this gate before any WMMA,
                // into distinct registers, so the loads pipeline and the three
                // accumulation WMMAs (D->C chain: no hazard NOPs) go back-to-back
                BF16Frag bw, bh0, bh1;
                const char* wp = (const char*)(sWih + n * 40);
                const char* hp = (const char*)(sWhh + n * 72);
                bw.q[0]  = *(const uint4*)(wp + lh * 16);
                bw.q[1]  = *(const uint4*)(wp + lh * 16 + 32);
                bh0.q[0] = *(const uint4*)(hp + lh * 16);
                bh0.q[1] = *(const uint4*)(hp + lh * 16 + 32);
                bh1.q[0] = *(const uint4*)(hp + 64 + lh * 16);
                bh1.q[1] = *(const uint4*)(hp + 64 + lh * 16 + 32);

                const float b = bv[g * 4 + j];
                v8f acc = (v8f){b, b, b, b, b, b, b, b};
                acc = __builtin_amdgcn_wmma_f32_16x16x32_bf16(
                          false, ax.v,  false, bw.v,  (short)0, acc, false, false);
                acc = __builtin_amdgcn_wmma_f32_16x16x32_bf16(
                          false, ah0.v, false, bh0.v, (short)0, acc, false, false);
                acc = __builtin_amdgcn_wmma_f32_16x16x32_bf16(
                          false, ah1.v, false, bh1.v, (short)0, acc, false, false);
                ga[g] = acc;
            }

            // ---- elementwise gate math (fp32), update c, write h back to LDS ----
            v8f cold = cc[j];
            #pragma unroll
            for (int r = 0; r < 8; ++r) {
                float iv = sigm(ga[0][r]);
                float fv = sigm(ga[1][r]);
                float gv = tanhx(ga[2][r]);
                float ov = sigm(ga[3][r]);
                float cn = fv * cold[r] + iv * gv;
                cold[r] = cn;
                float hv = ov * tanhx(cn);
                sHw[(r + 8 * lh) * 72 + j * 16 + mrow] = (__bf16)hv;
            }
            cc[j] = cold;
        }
    }

    // ---- FC on last hidden state: out[b, :] = h @ W_fc^T + b_fc ----
    if (lane < 16) {
        const __bf16* hp = sHw + lane * 72;
        const int brow = b0 + lane;
        #pragma unroll
        for (int o = 0; o < OUT_N; ++o) {
            float s = b_fc[o];
            const float* wf = W_fc + o * HID;
            #pragma unroll 8
            for (int k = 0; k < HID; ++k) s += (float)hp[k] * wf[k];
            out[brow * OUT_N + o] = s;
        }
    }
}

extern "C" void kernel_launch(void* const* d_in, const int* in_sizes, int n_in,
                              void* d_out, int out_size, void* d_ws, size_t ws_size,
                              hipStream_t stream) {
    const float* x    = (const float*)d_in[0];
    const float* W_ih = (const float*)d_in[1];
    const float* W_hh = (const float*)d_in[2];
    const float* b_ih = (const float*)d_in[3];
    const float* b_hh = (const float*)d_in[4];
    const float* W_fc = (const float*)d_in[5];
    const float* b_fc = (const float*)d_in[6];
    float* out = (float*)d_out;

    const int B = in_sizes[0] / (T_STEPS * INPUT_N);   // 16384
    dim3 grid(B / ROWS_PER_BLOCK);                     // 128 blocks
    dim3 block(256);                                   // 8 wave32
    lstm_fused_wmma<<<grid, block, SMEM_BYTES, stream>>>(
        x, W_ih, W_hh, b_ih, b_hh, W_fc, b_fc, out);
}
